// FitTorch_87857851007474
// MI455X (gfx1250) — matrix operand order repaired and checked by reading, not compiled, not run on hardware
//
#include <hip/hip_runtime.h>
#include <hip/hip_bf16.h>

#define ND      56
#define HID     64
#define NATOMS  50000
#define NCONF   100
#define APC     500
#define MROWS   (NATOMS * 20 * 3)   // 3,000,000 derivative rows
#define NTILES  (NATOMS / 16)       // 3125 exact

typedef __attribute__((ext_vector_type(16))) _Float16 v16h;
typedef __attribute__((ext_vector_type(8)))  float    v8f;
typedef float f4v __attribute__((ext_vector_type(4)));
typedef int   v4i __attribute__((ext_vector_type(4)));

#if __has_builtin(__builtin_amdgcn_global_load_async_to_lds_b128)
#define HAVE_ASYNC_LDS 1
typedef __attribute__((address_space(1))) v4i* gp_b128;   // global b128 ptr
typedef __attribute__((address_space(3))) v4i* lp_b128;   // LDS b128 ptr
#endif

// fast sigmoid: v_rcp_f32 instead of precise-division expansion
__device__ __forceinline__ float fast_sig(float z) {
    return __builtin_amdgcn_rcpf(1.f + __expf(-z));
}

// ---- WMMA fragment helpers (CDNA5 ISA 7.12.2 layouts, wave32) -------------

// A-matrix 16x32 f16: lane r in group g holds M=r; half i (v=i/2,p=i&1):
//   v<4 : K = g*8 + 2v + p        (K 0..7 / 8..15)
//   v>=4: K = 16 + g*8 + 2(v-4)+p (K 16..23 / 24..31)
__device__ __forceinline__ int a_kidx(int i, int g) {
    int v = i >> 1, p = i & 1;
    return (v < 4) ? (g * 8 + v * 2 + p) : (16 + g * 8 + (v - 4) * 2 + p);
}

__device__ __forceinline__ v16h loadA_g(const float* __restrict__ src, int ld,
                                        int m0, int k0, int kmax, int r, int g) {
    v16h a;
#pragma unroll
    for (int i = 0; i < 16; ++i) {
        int k = k0 + a_kidx(i, g);
        a[i] = (k < kmax) ? (_Float16)src[(m0 + r) * ld + k] : (_Float16)0.f;
    }
    return a;
}

__device__ __forceinline__ v16h loadA_l(const _Float16* t, int k0, int r, int g) {
    v16h a;
#pragma unroll
    for (int i = 0; i < 16; ++i)
        a[i] = t[r * 72 + k0 + a_kidx(i, g)];
    return a;
}

// B-matrix 32x16 f16: lane r in group g holds N=r; half i -> K = a_kidx(i, g)
__device__ __forceinline__ v16h loadB_g(const float* __restrict__ W, int ld,
                                        int k0, int n0, int kmax, int r, int g) {
    v16h b;
#pragma unroll
    for (int i = 0; i < 16; ++i) {
        int k = k0 + a_kidx(i, g);
        b[i] = (k < kmax) ? (_Float16)W[k * ld + n0 + r] : (_Float16)0.f;
    }
    return b;
}

// B fragment of W^T where W is row-major [nrows][ld]: B[k][n] = W[n][k]
__device__ __forceinline__ v16h loadBT_g(const float* __restrict__ W, int ld,
                                         int k0, int n0, int nmax, int r, int g) {
    v16h b;
    int n = n0 + r;
#pragma unroll
    for (int i = 0; i < 16; ++i) {
        int k = k0 + a_kidx(i, g);
        b[i] = (n < nmax) ? (_Float16)W[n * ld + k] : (_Float16)0.f;
    }
    return b;
}

#define WMMA(a, b, c) __builtin_amdgcn_wmma_f32_16x16x32_f16( \
        false, (a), false, (b), (short)0, (c), false, false)

// ---- kernel 0: zero outputs ------------------------------------------------
__global__ void fit_zero_kernel(float* __restrict__ p, int n) {
    int i = blockIdx.x * blockDim.x + threadIdx.x;
    if (i < n) p[i] = 0.f;
}

// ---- kernel 1: MLP forward + backward (WMMA), per 16-atom tile per wave ----
__global__ __launch_bounds__(256)
void fit_mlp_kernel(const float* __restrict__ x,
                    const float* __restrict__ W1, const float* __restrict__ b1,
                    const float* __restrict__ W2, const float* __restrict__ b2,
                    const float* __restrict__ W3, const float* __restrict__ b3,
                    float* __restrict__ eOut,      // [NCONF]
                    float* __restrict__ dEdD)      // [NATOMS*ND]
{
    __shared__ _Float16 lds[8][16][72];            // per-wave private act tile
    const int lane = threadIdx.x & 31;
    const int w    = threadIdx.x >> 5;
    const int r    = lane & 15;
    const int g    = lane >> 4;

    const int tile    = blockIdx.x * 8 + w;
    const bool active = (tile < NTILES);
    const int t       = active ? tile : (NTILES - 1); // clamp; dupes don't store
    const int atom0   = t * 16;
    _Float16* my = &lds[w][0][0];

    // ---------------- GEMM1: z1 = x @ W1 + b1 ; h1 = silu(z1) -------------
    v16h ax0 = loadA_g(x, ND, atom0, 0,  ND, r, g);
    v16h ax1 = loadA_g(x, ND, atom0, 32, ND, r, g);
    v8f  sp1[4];                                   // silu'(z1) in C-layout
    float ep[8];
#pragma unroll
    for (int j = 0; j < 8; ++j) ep[j] = 0.f;

#pragma unroll
    for (int nt = 0; nt < 4; ++nt) {
        const int n0 = nt * 16;
        v16h bb0 = loadB_g(W1, HID, 0,  n0, ND, r, g);
        v16h bb1 = loadB_g(W1, HID, 32, n0, ND, r, g);
        v8f c = {0.f, 0.f, 0.f, 0.f, 0.f, 0.f, 0.f, 0.f};
        c = WMMA(ax0, bb0, c);
        c = WMMA(ax1, bb1, c);
        const float bias = b1[n0 + r];
#pragma unroll
        for (int j = 0; j < 8; ++j) {
            float z  = c[j] + bias;
            float sg = fast_sig(z);
            sp1[nt][j] = sg * (1.f + z * (1.f - sg));
            my[(j + 8 * g) * 72 + n0 + r] = (_Float16)(z * sg);   // h1
        }
    }
    __syncthreads();

    // ---------------- GEMM2: z2 = h1 @ W2 + b2 ; energy ; dz2 -------------
    v16h ah0 = loadA_l(my, 0,  r, g);
    v16h ah1 = loadA_l(my, 32, r, g);
#pragma unroll
    for (int nt = 0; nt < 4; ++nt) {
        const int n0 = nt * 16;
        v16h bb0 = loadB_g(W2, HID, 0,  n0, HID, r, g);
        v16h bb1 = loadB_g(W2, HID, 32, n0, HID, r, g);
        v8f c = {0.f, 0.f, 0.f, 0.f, 0.f, 0.f, 0.f, 0.f};
        c = WMMA(ah0, bb0, c);
        c = WMMA(ah1, bb1, c);
        const float bias = b2[n0 + r];
        const float w3n  = W3[n0 + r];
#pragma unroll
        for (int j = 0; j < 8; ++j) {
            float z   = c[j] + bias;
            float sg  = fast_sig(z);
            float h2  = z * sg;
            ep[j]    += h2 * w3n;                                  // e partial
            float sp2 = sg * (1.f + z * (1.f - sg));
            my[(j + 8 * g) * 72 + n0 + r] = (_Float16)(w3n * sp2); // dz2
        }
    }
    __syncthreads();

    // per-atom energy: butterfly over the 16 N-lanes, scatter to config
    const float b3v = b3[0];
#pragma unroll
    for (int j = 0; j < 8; ++j) {
        float e = ep[j];
        e += __shfl_xor(e, 1, 16);
        e += __shfl_xor(e, 2, 16);
        e += __shfl_xor(e, 4, 16);
        e += __shfl_xor(e, 8, 16);
        if (active && r == j) {
            int m = atom0 + j + 8 * g;
            atomicAdd(&eOut[m / APC], e + b3v);
        }
    }

    // ---------------- GEMM3: dh1 = dz2 @ W2^T ; dz1 = dh1 * silu'(z1) -----
    v16h ad0 = loadA_l(my, 0,  r, g);
    v16h ad1 = loadA_l(my, 32, r, g);
#pragma unroll
    for (int nt = 0; nt < 4; ++nt) {
        const int n0 = nt * 16;
        v16h bb0 = loadBT_g(W2, HID, 0,  n0, HID, r, g);
        v16h bb1 = loadBT_g(W2, HID, 32, n0, HID, r, g);
        v8f c = {0.f, 0.f, 0.f, 0.f, 0.f, 0.f, 0.f, 0.f};
        c = WMMA(ad0, bb0, c);
        c = WMMA(ad1, bb1, c);
#pragma unroll
        for (int j = 0; j < 8; ++j)
            my[(j + 8 * g) * 72 + n0 + r] = (_Float16)(c[j] * sp1[nt][j]); // dz1
    }
    __syncthreads();

    // ---------------- GEMM4: dEdD = dz1 @ W1^T  (N bound = 56) ------------
    v16h af0 = loadA_l(my, 0,  r, g);
    v16h af1 = loadA_l(my, 32, r, g);
#pragma unroll
    for (int nt = 0; nt < 4; ++nt) {
        const int n0 = nt * 16;
        v16h bb0 = loadBT_g(W1, HID, 0,  n0, ND, r, g);
        v16h bb1 = loadBT_g(W1, HID, 32, n0, ND, r, g);
        v8f c = {0.f, 0.f, 0.f, 0.f, 0.f, 0.f, 0.f, 0.f};
        c = WMMA(af0, bb0, c);
        c = WMMA(af1, bb1, c);
        const int n = n0 + r;
        if (active && n < ND) {
#pragma unroll
            for (int j = 0; j < 8; ++j)
                dEdD[(size_t)(atom0 + j + 8 * g) * ND + n] = c[j];
        }
    }
}

// ---- kernel 2: force contraction + scatter (bandwidth-bound: 672 MB xd) ----
#define ROWF 60   // LDS row stride in floats (240B, breaks 224B bank cycle)

__global__ __launch_bounds__(256)
void fit_force_kernel(const float*  __restrict__ xd,
                      const int*    __restrict__ xd_indx,
                      const int*    __restrict__ unique_j,
                      const float*  __restrict__ dEdD,
                      float*        __restrict__ fOut)   // [3*NATOMS]
{
#ifdef HAVE_ASYNC_LDS
    // gfx1250 path: async-stage 32 rows/wave into LDS, dot out of LDS.
    __shared__ float stage[8][32 * ROWF];
    const int  lane    = threadIdx.x & 31;
    const int  w       = threadIdx.x >> 5;
    const long rowbase = ((long)blockIdx.x * 8 + w) * 32;   // 32 | MROWS exactly
    if (rowbase >= MROWS) return;
    const long m = rowbase + lane;

    const float* gsrc = xd + (size_t)m * ND;                // 224B row, 16B aligned
    float*       ldst = &stage[w][lane * ROWF];
#pragma unroll
    for (int q = 0; q < 14; ++q)
        __builtin_amdgcn_global_load_async_to_lds_b128(
            (gp_b128)(gsrc + q * 4), (lp_b128)(ldst + q * 4), 0, 0);

    const int atom  = xd_indx[3 * m];
    const int coord = xd_indx[3 * m + 2];
    const f4v* dr = (const f4v*)(dEdD + (size_t)atom * ND); // L2-resident gather

#if __has_builtin(__builtin_amdgcn_s_wait_asynccnt)
    __builtin_amdgcn_s_wait_asynccnt(0);
#else
    asm volatile("s_wait_asynccnt 0" ::: "memory");
#endif

    const f4v* lr = (const f4v*)ldst;
    float s0 = 0.f, s1 = 0.f;
#pragma unroll
    for (int q = 0; q < 14; q += 2) {
        f4v a = lr[q],     b = dr[q];
        s0 += a.x * b.x + a.y * b.y + a.z * b.z + a.w * b.w;
        f4v a2 = lr[q + 1], b2 = dr[q + 1];
        s1 += a2.x * b2.x + a2.y * b2.y + a2.z * b2.z + a2.w * b2.w;
    }
    atomicAdd(&fOut[unique_j[m] * 3 + coord], s0 + s1);
#else
    // fallback: direct streaming loads with non-temporal hint on xd
    int m = blockIdx.x * 256 + threadIdx.x;
    if (m >= MROWS) return;

    if (m + 4096 < MROWS)
        __builtin_prefetch(xd + (size_t)(m + 4096) * ND, 0, 1);

    const int atom  = xd_indx[3 * m];
    const int coord = xd_indx[3 * m + 2];
    const f4v* xr = (const f4v*)(xd   + (size_t)m    * ND);
    const f4v* dr = (const f4v*)(dEdD + (size_t)atom * ND);

    float s0 = 0.f, s1 = 0.f;
#pragma unroll
    for (int q = 0; q < 14; q += 2) {
        f4v a  = __builtin_nontemporal_load(xr + q);
        f4v b  = dr[q];
        s0 += a.x * b.x + a.y * b.y + a.z * b.z + a.w * b.w;
        f4v a2 = __builtin_nontemporal_load(xr + q + 1);
        f4v b2 = dr[q + 1];
        s1 += a2.x * b2.x + a2.y * b2.y + a2.z * b2.z + a2.w * b2.w;
    }
    atomicAdd(&fOut[unique_j[m] * 3 + coord], s0 + s1);
#endif
}

// ---- launcher --------------------------------------------------------------
extern "C" void kernel_launch(void* const* d_in, const int* in_sizes, int n_in,
                              void* d_out, int out_size, void* d_ws, size_t ws_size,
                              hipStream_t stream) {
    const float* x        = (const float*)d_in[0];
    const float* xd       = (const float*)d_in[1];
    // d_in[2] = indices (atom//APC, recomputed), d_in[3] = atoms_per_structure
    const int*   xd_indx  = (const int*)d_in[4];
    const int*   unique_j = (const int*)d_in[5];
    const float* W1 = (const float*)d_in[6];
    const float* b1 = (const float*)d_in[7];
    const float* W2 = (const float*)d_in[8];
    const float* b2 = (const float*)d_in[9];
    const float* W3 = (const float*)d_in[10];
    const float* b3 = (const float*)d_in[11];

    float* out    = (float*)d_out;              // [0:100)=energy, [100:150100)=forces
    float* dEdD   = (float*)d_ws;               // NATOMS*ND f32 = 11.2 MB
    float* energy = out;
    float* forces = out + NCONF;

    const int nOut = NCONF + 3 * NATOMS;        // 150100
    fit_zero_kernel<<<(nOut + 255) / 256, 256, 0, stream>>>(out, nOut);

    fit_mlp_kernel<<<(NTILES + 7) / 8, 256, 0, stream>>>(
        x, W1, b1, W2, b2, W3, b3, energy, dEdD);

    fit_force_kernel<<<(MROWS + 255) / 256, 256, 0, stream>>>(
        xd, xd_indx, unique_j, dEdD, forces);
}
